// LiteMLA_12635793785446
// MI455X (gfx1250) — matrix operand order repaired and checked by reading, not compiled, not run on hardware
//
#include <hip/hip_runtime.h>

typedef __attribute__((ext_vector_type(16))) __bf16 v16bf;
typedef __attribute__((ext_vector_type(8)))  __bf16 v8bf;
typedef __attribute__((ext_vector_type(4)))  __bf16 v4bf;
typedef __attribute__((ext_vector_type(2)))  __bf16 v2bf;
typedef __attribute__((ext_vector_type(8)))  float  v8f;
typedef __attribute__((ext_vector_type(4)))  unsigned int u32x4;
typedef __attribute__((ext_vector_type(8)))  int i32x8;
typedef __attribute__((ext_vector_type(4)))  int i32x4;

#define NPIX 16384  // H*W = 128*128

#if defined(__has_builtin)
#if __has_builtin(__builtin_amdgcn_tensor_load_to_lds) && __has_builtin(__builtin_amdgcn_s_wait_tensorcnt)
#define HAVE_TDM 1
#endif
#endif

// ---------------- WMMA fragment helpers (bf16, 16x16x32) ----------------
__device__ inline v16bf load_a_frag(const __bf16* tile, int mBase) {
    const int lane = threadIdx.x & 31;
    const int row  = mBase + (lane & 15);
    const int hi   = lane >> 4;
    const __bf16* p = tile + row * 32;
    v8bf lo = *(const v8bf*)(p + hi * 8);
    v8bf hh = *(const v8bf*)(p + 16 + hi * 8);
    v16bf r;
#pragma unroll
    for (int i = 0; i < 8; ++i) { r[i] = lo[i]; r[i + 8] = hh[i]; }
    return r;
}

__device__ inline v16bf load_b_frag(const __bf16* tileT, int nBase) {
    const int lane = threadIdx.x & 31;
    const int col  = nBase + (lane & 15);
    const int k0   = (lane >> 4) * 8;
    const __bf16* p = tileT + col * 32 + k0;
    v8bf a = *(const v8bf*)p;
    v8bf b = *(const v8bf*)(p + 16);
    v16bf r;
#pragma unroll
    for (int i = 0; i < 8; ++i) { r[i] = a[i]; r[i + 8] = b[i]; }
    return r;
}

// ---------------- K1: qkv = w_qkv(1536x512) * x(512xN), bf16 out ------------
__global__ __launch_bounds__(256) void qkv_gemm(const float* __restrict__ x,
                                                const float* __restrict__ wq,
                                                __bf16* __restrict__ qkv) {
    __shared__ __bf16 Wt[2][128 * 32];
    __shared__ __bf16 XT[2][128 * 32];
    const int n0 = blockIdx.x * 128;
    const int o0 = blockIdx.y * 128;
    const int b  = blockIdx.z;
    const int tid  = threadIdx.x;
    const int wave = tid >> 5;
    const int lane = tid & 31;
    v8f acc[8] = {};
    float4 wr[4], xr[4];

#pragma unroll
    for (int u = 0; u < 4; ++u) {
        const int s = tid + u * 256;
        wr[u] = *(const float4*)(wq + (size_t)(o0 + (s >> 3)) * 512 + ((s & 7) << 2));
        xr[u] = *(const float4*)(x + ((size_t)b * 512 + (s >> 5)) * NPIX + n0 + ((s & 31) << 2));
    }
#pragma unroll
    for (int u = 0; u < 4; ++u) {
        const int s = tid + u * 256;
        __bf16* p = &Wt[0][(s >> 3) * 32 + ((s & 7) << 2)];
        p[0] = (__bf16)wr[u].x; p[1] = (__bf16)wr[u].y; p[2] = (__bf16)wr[u].z; p[3] = (__bf16)wr[u].w;
        const int k = s >> 5, nq = (s & 31) << 2;
        XT[0][(nq + 0) * 32 + k] = (__bf16)xr[u].x;
        XT[0][(nq + 1) * 32 + k] = (__bf16)xr[u].y;
        XT[0][(nq + 2) * 32 + k] = (__bf16)xr[u].z;
        XT[0][(nq + 3) * 32 + k] = (__bf16)xr[u].w;
    }

    int cur = 0;
    for (int kk = 32; kk <= 512; kk += 32) {
        const bool more = (kk < 512);
        __syncthreads();
        if (more) {
#pragma unroll
            for (int u = 0; u < 4; ++u) {
                const int s = tid + u * 256;
                wr[u] = *(const float4*)(wq + (size_t)(o0 + (s >> 3)) * 512 + kk + ((s & 7) << 2));
                xr[u] = *(const float4*)(x + ((size_t)b * 512 + kk + (s >> 5)) * NPIX + n0 + ((s & 31) << 2));
            }
        }
        const __bf16* Wb = Wt[cur];
        const __bf16* Xb = XT[cur];
        const v16bf a = load_a_frag(Wb, wave * 16);
        v16bf bf[8];
#pragma unroll
        for (int j = 0; j < 8; ++j) bf[j] = load_b_frag(Xb, j * 16);   // all ds loads in flight
#pragma unroll
        for (int j = 0; j < 8; ++j)
            acc[j] = __builtin_amdgcn_wmma_f32_16x16x32_bf16(
                false, a, false, bf[j], (short)0, acc[j], false, false);
        if (more) {
            const int nb = cur ^ 1;
#pragma unroll
            for (int u = 0; u < 4; ++u) {
                const int s = tid + u * 256;
                __bf16* p = &Wt[nb][(s >> 3) * 32 + ((s & 7) << 2)];
                p[0] = (__bf16)wr[u].x; p[1] = (__bf16)wr[u].y; p[2] = (__bf16)wr[u].z; p[3] = (__bf16)wr[u].w;
                const int k = s >> 5, nq = (s & 31) << 2;
                XT[nb][(nq + 0) * 32 + k] = (__bf16)xr[u].x;
                XT[nb][(nq + 1) * 32 + k] = (__bf16)xr[u].y;
                XT[nb][(nq + 2) * 32 + k] = (__bf16)xr[u].z;
                XT[nb][(nq + 3) * 32 + k] = (__bf16)xr[u].w;
            }
        }
        cur ^= 1;
    }
    const int col  = lane & 15;
    const int rofs = (lane >> 4) * 8;
#pragma unroll
    for (int j = 0; j < 8; ++j)
#pragma unroll
        for (int r = 0; r < 8; ++r) {
            const int o = o0 + wave * 16 + rofs + r;
            qkv[((size_t)b * 1536 + o) * NPIX + n0 + j * 16 + col] = (__bf16)acc[j][r];
        }
}

// ---------------- K2: depthwise 5x5 (pad2) + grouped 8->8 pointwise ---------
__global__ __launch_bounds__(256) void dwpw(const __bf16* __restrict__ qkv,
                                            const float* __restrict__ wdw,
                                            const float* __restrict__ wpw,
                                            __bf16* __restrict__ y) {
    __shared__ __bf16 tile[8][36 * 36];
    __shared__ float  sdw[8 * 25];
    __shared__ float  spw[64];
    const int st = blockIdx.x, g = blockIdx.y, b = blockIdx.z;
    const int th0 = (st >> 2) * 32, tw0 = (st & 3) * 32;
    const int tid = threadIdx.x;
    for (int i = tid; i < 200; i += 256) sdw[i] = wdw[(size_t)(g * 8 + i / 25) * 25 + i % 25];
    for (int i = tid; i < 64;  i += 256) spw[i] = wpw[(size_t)(g * 8 + (i >> 3)) * 8 + (i & 7)];

#ifdef HAVE_TDM
    // Interior spatial tiles: one Tensor-Data-Mover 3D tile load (36x36x8 bf16)
    const bool interior = (th0 >= 32) && (th0 <= 64) && (tw0 >= 32) && (tw0 <= 64);
    if (interior) {
        if (tid < 32) {
            const __bf16* gsrc = qkv + ((size_t)b * 1536 + g * 8) * NPIX
                                     + (size_t)(th0 - 2) * 128 + (tw0 - 2);
            const unsigned long long ga = (unsigned long long)(uintptr_t)gsrc;
            const unsigned int lds = (unsigned int)(uintptr_t)(&tile[0][0]);
            u32x4 g0;
            g0[0] = 1u;
            g0[1] = lds;
            g0[2] = (unsigned int)ga;
            g0[3] = (unsigned int)((ga >> 32) & 0x01FFFFFFu) | (2u << 30);
            const unsigned int td0 = 1u << 30, td1 = 1u << 30;
            const unsigned int tl0 = 36, tl1 = 36, tl2 = 8;
            const unsigned long long s0 = 128ull;
            const unsigned long long s1 = (unsigned long long)NPIX;
            i32x8 g1;
            g1[0] = (int)(1u << 16);
            g1[1] = (int)((td0 & 0xFFFFu) << 16);
            g1[2] = (int)((td0 >> 16) | ((td1 & 0xFFFFu) << 16));
            g1[3] = (int)((td1 >> 16) | (tl0 << 16));
            g1[4] = (int)(tl1 | (tl2 << 16));
            g1[5] = (int)(unsigned int)s0;
            g1[6] = (int)((unsigned int)(s0 >> 32) | ((unsigned int)(s1 & 0xFFFFull) << 16));
            g1[7] = (int)(unsigned int)(s1 >> 16);
            i32x4 g2;
            g2[0] = (int)(1u << 30);
            g2[1] = 1;
            g2[2] = 0;
            g2[3] = 0;
            i32x4 g3 = {0, 0, 0, 0};
#if __clang_major__ >= 23
            i32x8 g4 = {0, 0, 0, 0, 0, 0, 0, 0};
            __builtin_amdgcn_tensor_load_to_lds(g0, g1, g2, g3, g4, 0);
#else
            __builtin_amdgcn_tensor_load_to_lds(g0, g1, g2, g3, 0);
#endif
            __builtin_amdgcn_s_wait_tensorcnt(0);
        }
    } else
#endif
    {
        for (int s = tid; s < 8 * 1296; s += 256) {
            const int i = s / 1296, rem = s % 1296;
            const int hy = rem / 36, hx = rem % 36;
            const int h = th0 + hy - 2, w = tw0 + hx - 2;
            __bf16 val = (__bf16)0.f;
            if (h >= 0 && h < 128 && w >= 0 && w < 128)
                val = qkv[((size_t)b * 1536 + g * 8 + i) * NPIX + h * 128 + w];
            tile[i][rem] = val;
        }
    }
    __syncthreads();

    // 4 adjacent pixels per thread; row segments reused across taps
    {
        const int p0 = tid << 2;
        const int ly = p0 >> 5, lx0 = p0 & 31;
        float t[4][8];
#pragma unroll
        for (int i = 0; i < 8; ++i) {
            float a0 = 0.f, a1 = 0.f, a2 = 0.f, a3 = 0.f;
#pragma unroll
            for (int dy = 0; dy < 5; ++dy) {
                float rowv[8];
#pragma unroll
                for (int c = 0; c < 8; ++c)
                    rowv[c] = (float)tile[i][(ly + dy) * 36 + lx0 + c];
#pragma unroll
                for (int dx = 0; dx < 5; ++dx) {
                    const float wgt = sdw[i * 25 + dy * 5 + dx];
                    a0 += rowv[dx + 0] * wgt;
                    a1 += rowv[dx + 1] * wgt;
                    a2 += rowv[dx + 2] * wgt;
                    a3 += rowv[dx + 3] * wgt;
                }
            }
            t[0][i] = a0; t[1][i] = a1; t[2][i] = a2; t[3][i] = a3;
        }
        const int h = th0 + ly, w0 = tw0 + lx0;
#pragma unroll
        for (int o = 0; o < 8; ++o) {
            v4bf ov;
#pragma unroll
            for (int j = 0; j < 4; ++j) {
                float a = 0.f;
#pragma unroll
                for (int i = 0; i < 8; ++i) a += spw[o * 8 + i] * t[j][i];
                ov[j] = (__bf16)a;
            }
            *(v4bf*)(y + ((size_t)b * 1536 + g * 8 + o) * NPIX + h * 128 + w0) = ov;
        }
    }
}

// head h<64 reads qkv channels h*24.., h>=64 reads y channels (h-64)*24..
__device__ inline const __bf16* head_base(const __bf16* qkv, const __bf16* y, int b, int h) {
    return (h < 64) ? qkv + ((size_t)b * 1536 + h * 24) * NPIX
                    : y   + ((size_t)b * 1536 + (h - 64) * 24) * NPIX;
}

// ---------------- K3a: vk[9][8] = sum_n v_pad[d,n]*relu(k[e,n]) -------------
__global__ __launch_bounds__(256) void att_vk(const __bf16* __restrict__ qkv,
                                              const __bf16* __restrict__ y,
                                              float* __restrict__ vk) {
    const int nc = blockIdx.x, h = blockIdx.y, b = blockIdx.z;
    const __bf16* base = head_base(qkv, y, b, h);
    float acc[72];
#pragma unroll
    for (int i = 0; i < 72; ++i) acc[i] = 0.f;
    const int tid = threadIdx.x;
#pragma unroll
    for (int u = 0; u < 2; ++u) {
        const int n2 = nc * 1024 + u * 512 + (tid << 1);   // 2 consecutive n
        float k0[8], k1[8], v0[8], v1[8];
#pragma unroll
        for (int e = 0; e < 8; ++e) {
            const v2bf t = *(const v2bf*)(base + (size_t)(8 + e) * NPIX + n2);
            const float a = (float)t[0], c = (float)t[1];
            k0[e] = a > 0.f ? a : 0.f;
            k1[e] = c > 0.f ? c : 0.f;
        }
#pragma unroll
        for (int d = 0; d < 8; ++d) {
            const v2bf t = *(const v2bf*)(base + (size_t)(16 + d) * NPIX + n2);
            v0[d] = (float)t[0]; v1[d] = (float)t[1];
        }
#pragma unroll
        for (int d = 0; d < 8; ++d)
#pragma unroll
            for (int e = 0; e < 8; ++e) acc[d * 8 + e] += v0[d] * k0[e] + v1[d] * k1[e];
#pragma unroll
        for (int e = 0; e < 8; ++e) acc[64 + e] += k0[e] + k1[e];
    }
    __shared__ float red[72];
    for (int i = tid; i < 72; i += 256) red[i] = 0.f;
    __syncthreads();
    const int lane = tid & 31;
#pragma unroll
    for (int i = 0; i < 72; ++i) {
        float v = acc[i];
        for (int off = 16; off > 0; off >>= 1) v += __shfl_xor(v, off);
        if (lane == 0) atomicAdd(&red[i], v);
    }
    __syncthreads();
    float* dst = vk + ((size_t)b * 128 + h) * 72;
    for (int i = tid; i < 72; i += 256) atomicAdd(&dst[i], red[i]);
}

// ---------------- K3b: out[d,n] = (vk . relu(q))_d / ((vk_ones . q) + eps) --
__global__ __launch_bounds__(256) void att_out(const __bf16* __restrict__ qkv,
                                               const __bf16* __restrict__ y,
                                               const float* __restrict__ vk,
                                               __bf16* __restrict__ att) {
    const int nc = blockIdx.x, h = blockIdx.y, b = blockIdx.z;
    const __bf16* base = head_base(qkv, y, b, h);
    __shared__ float svk[72];
    const int tid = threadIdx.x;
    for (int i = tid; i < 72; i += 256) svk[i] = vk[((size_t)b * 128 + h) * 72 + i];
    __syncthreads();
#pragma unroll
    for (int u = 0; u < 2; ++u) {
        const int n2 = nc * 1024 + u * 512 + (tid << 1);
        float q0[8], q1[8];
#pragma unroll
        for (int e = 0; e < 8; ++e) {
            const v2bf t = *(const v2bf*)(base + (size_t)e * NPIX + n2);
            const float a = (float)t[0], c = (float)t[1];
            q0[e] = a > 0.f ? a : 0.f;
            q1[e] = c > 0.f ? c : 0.f;
        }
        float den0 = 1e-15f, den1 = 1e-15f;
#pragma unroll
        for (int e = 0; e < 8; ++e) { den0 += svk[64 + e] * q0[e]; den1 += svk[64 + e] * q1[e]; }
        const float r0 = 1.f / den0, r1 = 1.f / den1;
#pragma unroll
        for (int d = 0; d < 8; ++d) {
            float n0v = 0.f, n1v = 0.f;
#pragma unroll
            for (int e = 0; e < 8; ++e) { n0v += svk[d * 8 + e] * q0[e]; n1v += svk[d * 8 + e] * q1[e]; }
            v2bf o;
            o[0] = (__bf16)(n0v * r0);
            o[1] = (__bf16)(n1v * r1);
            *(v2bf*)(att + ((size_t)b * 1024 + h * 8 + d) * NPIX + n2) = o;
        }
    }
}

// ---------------- K4: out = w_proj(512x1024) * att(1024xN), fused BN --------
__global__ __launch_bounds__(256) void proj_gemm_bn(const __bf16* __restrict__ att,
                                                    const float* __restrict__ wp,
                                                    const float* __restrict__ gamma,
                                                    const float* __restrict__ beta,
                                                    const float* __restrict__ mean,
                                                    const float* __restrict__ var,
                                                    float* __restrict__ out) {
    __shared__ __bf16 Wt[2][128 * 32];
    __shared__ __bf16 XT[2][128 * 32];
    __shared__ float  sInv[128], sBias[128];
    const int n0 = blockIdx.x * 128;
    const int o0 = blockIdx.y * 128;
    const int b  = blockIdx.z;
    const int tid  = threadIdx.x;
    const int wave = tid >> 5;
    const int lane = tid & 31;
    v8f acc[8] = {};
    float4 wr[4];
    v8bf  xrb[2];

    if (tid < 128) {
        const float iv = gamma[o0 + tid] * rsqrtf(var[o0 + tid] + 1e-5f);
        sInv[tid]  = iv;
        sBias[tid] = beta[o0 + tid] - mean[o0 + tid] * iv;
    }
#pragma unroll
    for (int u = 0; u < 4; ++u) {
        const int s = tid + u * 256;
        wr[u] = *(const float4*)(wp + (size_t)(o0 + (s >> 3)) * 1024 + ((s & 7) << 2));
    }
#pragma unroll
    for (int u = 0; u < 2; ++u) {
        const int s = tid + u * 256;
        xrb[u] = *(const v8bf*)(att + ((size_t)b * 1024 + (s >> 4)) * NPIX + n0 + ((s & 15) << 3));
    }
#pragma unroll
    for (int u = 0; u < 4; ++u) {
        const int s = tid + u * 256;
        __bf16* p = &Wt[0][(s >> 3) * 32 + ((s & 7) << 2)];
        p[0] = (__bf16)wr[u].x; p[1] = (__bf16)wr[u].y; p[2] = (__bf16)wr[u].z; p[3] = (__bf16)wr[u].w;
    }
#pragma unroll
    for (int u = 0; u < 2; ++u) {
        const int s = tid + u * 256;
        const int k = s >> 4, nq = (s & 15) << 3;
#pragma unroll
        for (int j = 0; j < 8; ++j) XT[0][(nq + j) * 32 + k] = xrb[u][j];
    }

    int cur = 0;
    for (int kk = 32; kk <= 1024; kk += 32) {
        const bool more = (kk < 1024);
        __syncthreads();
        if (more) {
#pragma unroll
            for (int u = 0; u < 4; ++u) {
                const int s = tid + u * 256;
                wr[u] = *(const float4*)(wp + (size_t)(o0 + (s >> 3)) * 1024 + kk + ((s & 7) << 2));
            }
#pragma unroll
            for (int u = 0; u < 2; ++u) {
                const int s = tid + u * 256;
                xrb[u] = *(const v8bf*)(att + ((size_t)b * 1024 + kk + (s >> 4)) * NPIX + n0 + ((s & 15) << 3));
            }
        }
        const __bf16* Wb = Wt[cur];
        const __bf16* Xb = XT[cur];
        const v16bf a = load_a_frag(Wb, wave * 16);
        v16bf bf[8];
#pragma unroll
        for (int j = 0; j < 8; ++j) bf[j] = load_b_frag(Xb, j * 16);
#pragma unroll
        for (int j = 0; j < 8; ++j)
            acc[j] = __builtin_amdgcn_wmma_f32_16x16x32_bf16(
                false, a, false, bf[j], (short)0, acc[j], false, false);
        if (more) {
            const int nb = cur ^ 1;
#pragma unroll
            for (int u = 0; u < 4; ++u) {
                const int s = tid + u * 256;
                __bf16* p = &Wt[nb][(s >> 3) * 32 + ((s & 7) << 2)];
                p[0] = (__bf16)wr[u].x; p[1] = (__bf16)wr[u].y; p[2] = (__bf16)wr[u].z; p[3] = (__bf16)wr[u].w;
            }
#pragma unroll
            for (int u = 0; u < 2; ++u) {
                const int s = tid + u * 256;
                const int k = s >> 4, nq = (s & 15) << 3;
#pragma unroll
                for (int j = 0; j < 8; ++j) XT[nb][(nq + j) * 32 + k] = xrb[u][j];
            }
        }
        cur ^= 1;
    }
    const int col  = lane & 15;
    const int rofs = (lane >> 4) * 8;
#pragma unroll
    for (int j = 0; j < 8; ++j)
#pragma unroll
        for (int r = 0; r < 8; ++r) {
            const int ol = wave * 16 + rofs + r;
            out[((size_t)b * 512 + o0 + ol) * NPIX + n0 + j * 16 + col]
                = acc[j][r] * sInv[ol] + sBias[ol];
        }
}

// ---------------- launch --------------------------------------------------
extern "C" void kernel_launch(void* const* d_in, const int* in_sizes, int n_in,
                              void* d_out, int out_size, void* d_ws, size_t ws_size,
                              hipStream_t stream) {
    const float* x     = (const float*)d_in[0];
    const float* wq    = (const float*)d_in[1];
    const float* wdw   = (const float*)d_in[2];
    const float* wpw   = (const float*)d_in[3];
    const float* wproj = (const float*)d_in[4];
    const float* gamma = (const float*)d_in[5];
    const float* beta  = (const float*)d_in[6];
    const float* mean  = (const float*)d_in[7];
    const float* var   = (const float*)d_in[8];
    float* out = (float*)d_out;

    char* ws = (char*)d_ws;
    __bf16* qkv = (__bf16*)(ws);                      // 96 MiB
    __bf16* y   = (__bf16*)(ws + (size_t)100663296);  // 96 MiB
    __bf16* att = (__bf16*)(ws + (size_t)201326592);  // 64 MiB
    float*  vk  = (float*)(ws + (size_t)268435456);   // 2*128*72 f32

    hipMemsetAsync(vk, 0, (size_t)2 * 128 * 72 * sizeof(float), stream);

    qkv_gemm<<<dim3(128, 12, 2), 256, 0, stream>>>(x, wq, qkv);
    dwpw<<<dim3(16, 192, 2), 256, 0, stream>>>(qkv, wdw, wpw, y);
    att_vk<<<dim3(16, 128, 2), 256, 0, stream>>>(qkv, y, vk);
    att_out<<<dim3(16, 128, 2), 256, 0, stream>>>(qkv, y, vk, att);
    proj_gemm_bn<<<dim3(128, 4, 2), 256, 0, stream>>>(att, wproj, gamma, beta, mean, var, out);
}
